// OCAB_26731876450666
// MI455X (gfx1250) — matrix operand order repaired and checked by reading, not compiled, not run on hardware
//
#include <hip/hip_runtime.h>
#include <hip/hip_bf16.h>
#include <cstdint>

typedef __attribute__((ext_vector_type(8)))  float        v8f;
typedef __attribute__((ext_vector_type(16))) __bf16       v16bf;
typedef __attribute__((ext_vector_type(4)))  unsigned int u32x4;
typedef __attribute__((ext_vector_type(8)))  int          i32x8;
typedef __attribute__((ext_vector_type(4)))  int          i32x4;

#ifndef __has_builtin
#define __has_builtin(x) 0
#endif
#if defined(__AMDGCN__) && __has_builtin(__builtin_amdgcn_tensor_load_to_lds) && \
    __has_builtin(__builtin_amdgcn_s_wait_tensorcnt)
#define HAVE_TDM 1
#else
#define HAVE_TDM 0
#endif

#define DIM   180
#define DIMP  192
#define HID   360
#define HIDP  384
#define HEADS 6
#define HD    30
#define NKEY  576   // 24*24
#define L     65536 // 256*256

// ---------- fragment loads ----------
__device__ __forceinline__ v16bf load_a_frag(const __bf16* p) {
    // A-matrix 16x32 bf16: lane<16 -> K {0..7,16..23}; lane>=16 -> {8..15,24..31}
    union { v16bf v; uint4 u[2]; } f;
    f.u[0] = *(const uint4*)(p);
    f.u[1] = *(const uint4*)(p + 16);
    return f.v;
}
__device__ __forceinline__ v16bf load_b_frag(const __bf16* p) {
    // B-matrix 32x16 bf16: lane<16 -> K {0..7,16..23}; lane>=16 -> {8..15,24..31}
    union { v16bf v; uint4 u[2]; } f;
    f.u[0] = *(const uint4*)(p);
    f.u[1] = *(const uint4*)(p + 16);
    return f.v;
}
__device__ __forceinline__ int f2ord(float f) { int i = __float_as_int(f); return (i >= 0) ? i : (i ^ 0x7fffffff); }
__device__ __forceinline__ float ord2f(int i) { int j = (i >= 0) ? i : (i ^ 0x7fffffff); return __int_as_float(j); }

#if HAVE_TDM
// TDM: DMA one [NP rows][32 elem] bf16 chunk of Bt (row stride KP) into LDS.
// D# per CDNA5 ISA ch.8: group0 {count=1, lds_addr, global_addr, type=2},
// group1 {data_size=2B, dim0=KP, dim1=NP, tile0=32, tile1=NP, stride0=KP}.
template <int KP, int NP>
__device__ __forceinline__ void tdm_stage_b(const __bf16* Bt, int kt, unsigned lds_off) {
    unsigned long long ga = (unsigned long long)(uintptr_t)Bt + (unsigned long long)kt * 64ull;
    u32x4 g0 = { 1u, lds_off, (unsigned)(ga & 0xffffffffull),
                 (unsigned)((ga >> 32) & 0x01ffffffull) | 0x80000000u };
    i32x8 g1 = { 0x10000,                                        // data_size = 2 bytes
                 (KP & 0xffff) << 16,                            // tensor_dim0 lo
                 ((KP >> 16) & 0xffff) | ((NP & 0xffff) << 16),  // dim0 hi | dim1 lo
                 ((NP >> 16) & 0xffff) | (32 << 16),             // dim1 hi | tile_dim0
                 (NP & 0xffff),                                  // tile_dim1 (tile_dim2 = 0)
                 KP,                                             // tensor_dim0_stride lo
                 0, 0 };
    i32x4 gz = { 0, 0, 0, 0 };
#if __clang_major__ >= 23
    i32x8 gz8 = { 0, 0, 0, 0, 0, 0, 0, 0 };
    __builtin_amdgcn_tensor_load_to_lds(g0, g1, gz, gz, gz8, 0);
#else
    __builtin_amdgcn_tensor_load_to_lds(g0, g1, gz, gz, 0);
#endif
}
#endif

// ---------- weight prep: Bt[n][k] = W[k][n], bf16, zero-padded ----------
__global__ void prep_w_kernel(const float* __restrict__ w, __bf16* __restrict__ bt,
                              int K, int N, int KP, int NP) {
    int idx = blockIdx.x * 256 + threadIdx.x;
    if (idx >= NP * KP) return;
    int n = idx / KP, k = idx % KP;
    float v = (n < N && k < K) ? w[k * N + n] : 0.f;
    bt[idx] = (__bf16)v;
}

// ---------- layernorm: f32 [M][180] -> bf16 [M][192] (pad zero) ----------
__global__ void ln_kernel(const float* __restrict__ x, const float* __restrict__ g,
                          const float* __restrict__ bta, __bf16* __restrict__ out, int M) {
    int gid  = blockIdx.x * blockDim.x + threadIdx.x;
    int row  = gid >> 5;
    int lane = gid & 31;
    if (row >= M) return;
    const float* xr = x + (size_t)row * DIM;
    float vals[6]; float s = 0.f;
#pragma unroll
    for (int i = 0; i < 6; i++) { int c = lane + i * 32; float v = (c < DIM) ? xr[c] : 0.f; vals[i] = v; s += v; }
#pragma unroll
    for (int off = 16; off >= 1; off >>= 1) s += __shfl_xor(s, off, 32);
    float mu = s * (1.f / DIM);
    float vs = 0.f;
#pragma unroll
    for (int i = 0; i < 6; i++) { int c = lane + i * 32; float d = (c < DIM) ? (vals[i] - mu) : 0.f; vals[i] = d; vs += d * d; }
#pragma unroll
    for (int off = 16; off >= 1; off >>= 1) vs += __shfl_xor(vs, off, 32);
    float rstd = rsqrtf(vs * (1.f / DIM) + 1e-5f);
    __bf16* orow = out + (size_t)row * DIMP;
#pragma unroll
    for (int i = 0; i < 6; i++) {
        int c = lane + i * 32;
        if (c < DIM)       orow[c] = (__bf16)(vals[i] * rstd * g[c] + bta[c]);
        else if (c < DIMP) orow[c] = (__bf16)0.f;
    }
}

// ---------- generic WMMA GEMM: C = A[M][KP](bf16) * Bt[NP][KP](bf16) + epilogue ----------
// B panel staged through LDS: TDM double-buffered DMA (tensor_load_to_lds) when
// available, cooperative vector-copy otherwise.
// EPI 0: +bias -> bf16 out[M][NP] (pads -> 0); EPI 1: gelu(+bias) -> bf16;
// EPI 2: +bias + res[M][N] -> f32 out[M][N]
template <int KP, int NP, int EPI>
__launch_bounds__(256)
__global__ void gemm_bf16_kernel(const __bf16* __restrict__ A, const __bf16* __restrict__ Bt,
                                 const float* __restrict__ bias, const float* __restrict__ res,
                                 float* __restrict__ outf, __bf16* __restrict__ outb, int N) {
    constexpr int NT = NP / 16;
    constexpr int KT = KP / 32;
    __shared__ __bf16 ldsB[2][NP * 32];

    const int wave  = threadIdx.x >> 5;
    const int lane  = threadIdx.x & 31;
    const int rowbase = blockIdx.x * 128 + wave * 16;
    const int rA    = lane & 15;
    const int koffA = (lane >> 4) * 8;
    const int koffB = (lane >> 4) * 8;

    v8f acc[NT] = {};

#if HAVE_TDM
    if (wave == 0) {
        tdm_stage_b<KP, NP>(Bt, 0, (unsigned)(uintptr_t)&ldsB[0][0]);
        __builtin_amdgcn_s_wait_tensorcnt(0);
    }
    __syncthreads();
    for (int kt = 0; kt < KT; ++kt) {
        const __bf16* cur = &ldsB[kt & 1][0];
        if (wave == 0 && kt + 1 < KT)
            tdm_stage_b<KP, NP>(Bt, kt + 1, (unsigned)(uintptr_t)&ldsB[(kt + 1) & 1][0]);
        const __bf16* ap = A + (size_t)(rowbase + rA) * KP + kt * 32 + koffA;
        if (kt + 1 < KT) __builtin_prefetch((const void*)(ap + 32), 0, 1);
        v16bf af = load_a_frag(ap);
#pragma unroll
        for (int nt = 0; nt < NT; ++nt) {
            v16bf bfr = load_b_frag(cur + (nt * 16 + rA) * 32 + koffB);
            acc[nt] = __builtin_amdgcn_wmma_f32_16x16x32_bf16(false, af, false, bfr,
                                                              (short)0, acc[nt], false, false);
        }
        if (wave == 0 && kt + 1 < KT) __builtin_amdgcn_s_wait_tensorcnt(0);
        __syncthreads();
    }
#else
    for (int kt = 0; kt < KT; ++kt) {
        // cooperative stage of B chunk kt: rows NP x 32 elems (4 x uint4 per row)
        const __bf16* src = Bt + kt * 32;
        for (int i = threadIdx.x; i < NP * 4; i += 256) {
            int row = i >> 2, part = i & 3;
            *(uint4*)&ldsB[0][row * 32 + part * 8] =
                *(const uint4*)(src + (size_t)row * KP + part * 8);
        }
        __syncthreads();
        const __bf16* cur = &ldsB[0][0];
        const __bf16* ap = A + (size_t)(rowbase + rA) * KP + kt * 32 + koffA;
        if (kt + 1 < KT) __builtin_prefetch((const void*)(ap + 32), 0, 1);
        v16bf af = load_a_frag(ap);
#pragma unroll
        for (int nt = 0; nt < NT; ++nt) {
            v16bf bfr = load_b_frag(cur + (nt * 16 + rA) * 32 + koffB);
            acc[nt] = __builtin_amdgcn_wmma_f32_16x16x32_bf16(false, af, false, bfr,
                                                              (short)0, acc[nt], false, false);
        }
        __syncthreads();
    }
#endif

    const int rowoff = (lane >> 4) << 3;
#pragma unroll
    for (int nt = 0; nt < NT; ++nt) {
        int col = nt * 16 + (lane & 15);
        float bsv = (col < N) ? bias[col] : 0.f;
#pragma unroll
        for (int r = 0; r < 8; ++r) {
            int row = rowbase + rowoff + r;
            float v = acc[nt][r] + bsv;
            if (EPI == 0) {
                outb[(size_t)row * NP + col] = (col < N) ? (__bf16)v : (__bf16)0.f;
            } else if (EPI == 1) {
                float gv = 0.5f * v * (1.f + erff(v * 0.70710678118654752f));
                outb[(size_t)row * NP + col] = (col < N) ? (__bf16)gv : (__bf16)0.f;
            } else {
                if (col < N) outf[(size_t)row * N + col] = v + res[(size_t)row * N + col];
            }
        }
    }
}

// ---------- attention: one block = (window, head, 32-query chunk) ----------
#define SM_K   0
#define SM_V   36864
#define SM_O   73728
#define SM_MAX 77824
#define SM_SUM 77952
#define SM_SZ  78080

__launch_bounds__(256)
__global__ void attn_kernel(const __bf16* __restrict__ Q, const __bf16* __restrict__ KV,
                            const int* __restrict__ rpi, const float* __restrict__ rpb,
                            __bf16* __restrict__ out) {
    extern __shared__ char smem[];
    __bf16* kbuf    = (__bf16*)(smem + SM_K);
    __bf16* vT      = (__bf16*)(smem + SM_V);
    float*  obuf    = (float*)(smem + SM_O);
    int*    rowmaxI = (int*)(smem + SM_MAX);
    float*  rowsum  = (float*)(smem + SM_SUM);

    const int win = blockIdx.x, head = blockIdx.y, chunk = blockIdx.z;
    const int wh = win >> 4, ww = win & 15;
    const int tid = threadIdx.x;
    const int lane = tid & 31, wave = tid >> 5;
    const int cbase = head * HD;

    // phase 1: init reductions + gather K,V (d padded 30->32 with zeros)
    if (tid < 32) { rowmaxI[tid] = f2ord(-3.4e38f); rowsum[tid] = 0.f; }
    for (int i = tid; i < 1024; i += 256) obuf[i] = 0.f;
    for (int i = 0; i < 72; ++i) {
        int idx = i * 256 + tid;            // 576*32 = 18432 elements
        int p = idx >> 5, d = idx & 31;
        int r = p / 24, cc = p % 24;
        int ky = wh * 16 - 4 + r, kx = ww * 16 - 4 + cc;
        float kvv = 0.f, vvv = 0.f;
        if (d < HD && ky >= 0 && ky < 256 && kx >= 0 && kx < 256) {
            const __bf16* src = KV + (size_t)(ky * 256 + kx) * HIDP + cbase + d;
            kvv = (float)src[0];
            vvv = (float)src[DIM];
        }
        kbuf[p * 32 + d] = (__bf16)kvv;
        vT[d * NKEY + p] = (__bf16)vvv;
    }
    __syncthreads();

    // phase 2: S = scale * Q K^T + bias  (wave w: M-tile w&1, 9 N-tiles)
    const int mt = wave & 1;
    const int ntbase = (wave >> 1) * 9;
    const int rA = lane & 15;
    const int koffA = (lane >> 4) * 8;
    const int koffB = (lane >> 4) * 8;
    const int rowoff = (lane >> 4) << 3;

    int qla = chunk * 32 + mt * 16 + rA;
    int qy = qla >> 4, qx = qla & 15;
    const __bf16* qp = Q + (size_t)((wh * 16 + qy) * 256 + ww * 16 + qx) * DIMP + cbase + koffA;
    v16bf af = load_a_frag(qp);

    v8f acc[9] = {};
#pragma unroll
    for (int j = 0; j < 9; ++j) {
        const __bf16* bp = kbuf + (size_t)((ntbase + j) * 16 + rA) * 32 + koffB;
        v16bf bfr = load_b_frag(bp);
        acc[j] = __builtin_amdgcn_wmma_f32_16x16x32_bf16(false, af, false, bfr,
                                                         (short)0, acc[j], false, false);
    }
    const float scale = 0.1825741858f; // 30^-0.5
    const int qlc = chunk * 32 + mt * 16 + rowoff;
#pragma unroll
    for (int j = 0; j < 9; ++j) {
        int coln = (ntbase + j) * 16 + (lane & 15);
#pragma unroll
        for (int r = 0; r < 8; ++r) {
            int ql = qlc + r;
            float b = rpb[rpi[ql * NKEY + coln] * HEADS + head];
            acc[j][r] = acc[j][r] * scale + b;
        }
    }
    __syncthreads();   // all waves done reading kbuf

    // phase 3: row max (LDS int-ordered atomic max)
#pragma unroll
    for (int r = 0; r < 8; ++r) {
        float m = -3.4e38f;
#pragma unroll
        for (int j = 0; j < 9; ++j) m = fmaxf(m, acc[j][r]);
        atomicMax(&rowmaxI[mt * 16 + rowoff + r], f2ord(m));
    }
    __syncthreads();

    // phase 4: exp, write P (unnormalized) over kbuf, accumulate row sums
    __bf16* pbuf = kbuf;   // [32][576]
#pragma unroll
    for (int r = 0; r < 8; ++r) {
        int rl = mt * 16 + rowoff + r;
        float rm = ord2f(rowmaxI[rl]);
        float s = 0.f;
#pragma unroll
        for (int j = 0; j < 9; ++j) {
            int coln = (ntbase + j) * 16 + (lane & 15);
            float e = __expf(acc[j][r] - rm);
            s += e;
            pbuf[(size_t)rl * NKEY + coln] = (__bf16)e;
        }
        atomicAdd(&rowsum[rl], s);
    }
    __syncthreads();

    // phase 5: O = P V  (waves split 2x2 output tiles x 2 K-halves, reduce via LDS)
    {
        int mt2 = (wave >> 1) & 1;
        int nt2 = wave & 1;
        int khalf = wave >> 2;
        v8f oacc = {};
#pragma unroll
        for (int j = 0; j < 9; ++j) {
            int kt = khalf * 9 + j;
            const __bf16* ap2 = pbuf + (size_t)(mt2 * 16 + rA) * NKEY + kt * 32 + koffA;
            v16bf a2 = load_a_frag(ap2);
            const __bf16* bp2 = vT + (size_t)(nt2 * 16 + rA) * NKEY + kt * 32 + koffB;
            v16bf b2 = load_b_frag(bp2);
            oacc = __builtin_amdgcn_wmma_f32_16x16x32_bf16(false, a2, false, b2,
                                                           (short)0, oacc, false, false);
        }
#pragma unroll
        for (int r = 0; r < 8; ++r)
            atomicAdd(&obuf[(mt2 * 16 + rowoff + r) * 32 + nt2 * 16 + (lane & 15)], oacc[r]);
    }
    __syncthreads();

    // phase 6: normalize and scatter to [pixel][head*30+d] (bf16, padded layout)
    for (int i = tid; i < 1024; i += 256) {
        int rl = i >> 5, d = i & 31;
        if (d < HD) {
            int ql = chunk * 32 + rl;
            int oy = ql >> 4, ox = ql & 15;
            size_t pix = (size_t)((wh * 16 + oy) * 256 + ww * 16 + ox);
            out[pix * DIMP + cbase + d] = (__bf16)(obuf[i] / rowsum[rl]);
        }
    }
}

// ---------- launch ----------
extern "C" void kernel_launch(void* const* d_in, const int* in_sizes, int n_in,
                              void* d_out, int out_size, void* d_ws, size_t ws_size,
                              hipStream_t stream) {
    (void)in_sizes; (void)n_in; (void)out_size; (void)ws_size;
    const float* x      = (const float*)d_in[0];
    const int*   rpi    = (const int*)d_in[1];
    const float* n1g    = (const float*)d_in[4];
    const float* n1b    = (const float*)d_in[5];
    const float* q_w    = (const float*)d_in[6];
    const float* q_b    = (const float*)d_in[7];
    const float* kv_w   = (const float*)d_in[8];
    const float* kv_b   = (const float*)d_in[9];
    const float* rpb    = (const float*)d_in[10];
    const float* proj_w = (const float*)d_in[11];
    const float* proj_b = (const float*)d_in[12];
    const float* n2g    = (const float*)d_in[13];
    const float* n2b    = (const float*)d_in[14];
    const float* mlp_w1 = (const float*)d_in[15];
    const float* mlp_b1 = (const float*)d_in[16];
    const float* mlp_w2 = (const float*)d_in[17];
    const float* mlp_b2 = (const float*)d_in[18];

    char* ws = (char*)d_ws;
    size_t off = 0;
    auto alloc = [&](size_t bytes) { char* p = ws + off; off += (bytes + 255) & ~(size_t)255; return p; };
    __bf16* btq    = (__bf16*)alloc((size_t)DIMP * DIMP * 2);
    __bf16* btkv   = (__bf16*)alloc((size_t)HIDP * DIMP * 2);
    __bf16* btproj = (__bf16*)alloc((size_t)DIMP * DIMP * 2);
    __bf16* btm1   = (__bf16*)alloc((size_t)HIDP * DIMP * 2);
    __bf16* btm2   = (__bf16*)alloc((size_t)DIMP * HIDP * 2);
    __bf16* xn     = (__bf16*)alloc((size_t)L * DIMP * 2);   // reused for xn2
    __bf16* qbuf   = (__bf16*)alloc((size_t)L * DIMP * 2);
    __bf16* kvbuf  = (__bf16*)alloc((size_t)L * HIDP * 2);   // reused for MLP hidden
    __bf16* abuf   = (__bf16*)alloc((size_t)L * DIMP * 2);
    float*  xo     = (float*)alloc((size_t)L * DIM * 4);

    // weight prep (bf16, transposed [N][K], padded)
    prep_w_kernel<<<(DIMP * DIMP + 255) / 256, 256, 0, stream>>>(q_w,    btq,    DIM, DIM, DIMP, DIMP);
    prep_w_kernel<<<(HIDP * DIMP + 255) / 256, 256, 0, stream>>>(kv_w,   btkv,   DIM, HID, DIMP, HIDP);
    prep_w_kernel<<<(DIMP * DIMP + 255) / 256, 256, 0, stream>>>(proj_w, btproj, DIM, DIM, DIMP, DIMP);
    prep_w_kernel<<<(HIDP * DIMP + 255) / 256, 256, 0, stream>>>(mlp_w1, btm1,   DIM, HID, DIMP, HIDP);
    prep_w_kernel<<<(DIMP * HIDP + 255) / 256, 256, 0, stream>>>(mlp_w2, btm2,   HID, DIM, HIDP, DIMP);

    // LN1
    ln_kernel<<<L / 8, 256, 0, stream>>>(x, n1g, n1b, xn, L);
    // Q = xn @ q_w + q_b ; KV = xn @ kv_w + kv_b
    gemm_bf16_kernel<DIMP, DIMP, 0><<<L / 128, 256, 0, stream>>>(xn, btq,  q_b,  nullptr, nullptr, qbuf,  DIM);
    gemm_bf16_kernel<DIMP, HIDP, 0><<<L / 128, 256, 0, stream>>>(xn, btkv, kv_b, nullptr, nullptr, kvbuf, HID);
    // attention output buffer (pad columns must be zero for proj GEMM)
    hipMemsetAsync(abuf, 0, (size_t)L * DIMP * 2, stream);
    attn_kernel<<<dim3(256, HEADS, 8), 256, SM_SZ, stream>>>(qbuf, kvbuf, rpi, rpb, abuf);
    // xo = attn @ proj_w + proj_b + x   (f32)
    gemm_bf16_kernel<DIMP, DIMP, 2><<<L / 128, 256, 0, stream>>>(abuf, btproj, proj_b, x, xo, nullptr, DIM);
    // LN2
    ln_kernel<<<L / 8, 256, 0, stream>>>(xo, n2g, n2b, xn, L);
    // h = gelu(xn2 @ mlp_w1 + mlp_b1)
    gemm_bf16_kernel<DIMP, HIDP, 1><<<L / 128, 256, 0, stream>>>(xn, btm1, mlp_b1, nullptr, nullptr, kvbuf, HID);
    // out = h @ mlp_w2 + mlp_b2 + xo   (f32 -> d_out)
    gemm_bf16_kernel<HIDP, DIMP, 2><<<L / 128, 256, 0, stream>>>(kvbuf, btm2, mlp_b2, xo, (float*)d_out, nullptr, DIM);
}
